// MultiHeadSelfAttention_75299366633636
// MI455X (gfx1250) — compile-verified, matrix operand order checked by reading, not run on hardware
//
#include <hip/hip_runtime.h>
#include <cmath>
#include <cstdint>

// ---------------------------------------------------------------------------
// Multi-head causal self-attention for MI455X (gfx1250, wave32, WMMA).
// b=2, s=2048, d_model=768, heads=12, dk=64.  bf16 WMMA w/ f32 accumulate.
// B operands (weights, K, V^T) flow global -> (async DMA) -> LDS -> ds_load -> WMMA (A rows from global),
// double-buffered on ASYNCcnt so DMA overlaps matrix math.
// ---------------------------------------------------------------------------

#define S_LEN   2048
#define DMODEL  768
#define NHEADS  12
#define DK      64
#define NBATCH  2
#define NROWS   (NBATCH * S_LEN)   // 4096
#define KSTEPS  (DMODEL / 32)      // 24

typedef __attribute__((ext_vector_type(16))) __bf16 v16bf;
typedef __attribute__((ext_vector_type(8)))  __bf16 v8bf;
typedef __attribute__((ext_vector_type(4)))  __bf16 v4bf;
typedef __attribute__((ext_vector_type(8)))  float  v8f;
typedef __attribute__((ext_vector_type(4)))  float  v4f;

// Load one 16-element A/B fragment for v_wmma_f32_16x16x32_bf16.
// Per ISA 7.12.2: lane (r = lane&15, kh = lane>>4) holds
//   elements 0..7  = row[kh*8   .. kh*8+7]
//   elements 8..15 = row[16+kh*8 .. 16+kh*8+7]
__device__ __forceinline__ v16bf load_frag(const __bf16* row, int kh) {
  v8bf lo = *(const v8bf*)(row + kh * 8);
  v8bf hi = *(const v8bf*)(row + 16 + kh * 8);
  v16bf r;
#pragma unroll
  for (int i = 0; i < 8; ++i) { r[i] = lo[i]; r[i + 8] = hi[i]; }
  return r;
}

__device__ __forceinline__ v8f wmma_bf16(v16bf a, v16bf b, v8f c) {
  return __builtin_amdgcn_wmma_f32_16x16x32_bf16(
      /*neg_a=*/false, a, /*neg_b=*/false, b,
      /*c_mod=*/(short)0, c, /*reuse_a=*/false, /*reuse_b=*/false);
}

// Async DMA: 16 bytes per lane, global -> LDS, tracked by ASYNCcnt.
__device__ __forceinline__ void async_copy_b128(const __bf16* gsrc, __bf16* ldst) {
  const uint32_t lds_addr = (uint32_t)(uintptr_t)ldst;   // flat->LDS: addr[31:0]
  const uint64_t gaddr    = (uint64_t)(uintptr_t)gsrc;
  asm volatile("global_load_async_to_lds_b128 %0, %1, off"
               :: "v"(lds_addr), "v"(gaddr) : "memory");
}

// ---------------------------------------------------------------------------
// fp32 -> bf16 conversion (4-wide)
// ---------------------------------------------------------------------------
__global__ void __launch_bounds__(256)
cvt_bf16_kernel(const float* __restrict__ src, __bf16* __restrict__ dst, int n4) {
  int i = blockIdx.x * blockDim.x + threadIdx.x;
  if (i < n4) {
    v4f v = *(const v4f*)(src + i * 4);
    v4bf o;
#pragma unroll
    for (int j = 0; j < 4; ++j) o[j] = (__bf16)v[j];
    *(v4bf*)(dst + i * 4) = o;
  }
}

// ---------------------------------------------------------------------------
// Fused QKV projection:  y = x @ W^T   (M=4096, N=768, K=768)
// grid = (M/128, N/64, 3), block = 128 (4 waves, each a 32x64 output tile).
// Weight tile (64 rows x 32 K = 4KB) async-DMA'd into double-buffered LDS and
// shared by all 4 waves; copy of step i+1 overlaps the WMMAs of step i.
// Q stored (b,h,s,dk) scaled by 1/8; K stored (b,h,s,dk); V stored (b,h,dk,s).
// ---------------------------------------------------------------------------
__global__ void __launch_bounds__(128)
qkv_proj_kernel(const __bf16* __restrict__ Xb,
                const __bf16* __restrict__ Wq,
                const __bf16* __restrict__ Wk,
                const __bf16* __restrict__ Wv,
                __bf16* __restrict__ Qb,
                __bf16* __restrict__ Kb,
                __bf16* __restrict__ Vtb) {
  __shared__ __align__(16) __bf16 bshm[2][64][32];   // [buf][n_row][k]

  const int tid  = threadIdx.x;
  const int lane = tid & 31;
  const int wid  = tid >> 5;
  const int kh   = lane >> 4;
  const int lm   = lane & 15;

  const int m0 = blockIdx.x * 128 + wid * 32;  // row tile in (b*s), 32 rows/wave
  const int n0 = blockIdx.y * 64;              // col tile in d_out
  const int which = blockIdx.z;                // 0=Q 1=K 2=V
  const __bf16* W = (which == 0) ? Wq : (which == 1) ? Wk : Wv;

  // 128 threads DMA the 4KB weight tile as 256 16B chunks (2 per thread).
  const int r0 = tid >> 2,         p0 = tid & 3;
  const int r1 = (tid + 128) >> 2, p1 = tid & 3;   // second chunk, rows 32..63
  auto issue_b_copy = [&](int buf, int kk) {
    async_copy_b128(W + (size_t)(n0 + r0) * DMODEL + kk + p0 * 8, &bshm[buf][r0][p0 * 8]);
    async_copy_b128(W + (size_t)(n0 + r1) * DMODEL + kk + p1 * 8, &bshm[buf][r1][p1 * 8]);
  };

  const __bf16* arow0 = Xb + (size_t)(m0 + lm) * DMODEL;
  const __bf16* arow1 = Xb + (size_t)(m0 + 16 + lm) * DMODEL;

  v8f acc[8] = {};   // [ms*4 + t]

  issue_b_copy(0, 0);
#pragma unroll 1   // keep single body: no accumulator rotation / WMMA reg churn
  for (int i = 0; i < KSTEPS; ++i) {
    const int kk = i * 32;
    if (i + 1 < KSTEPS) {
      issue_b_copy((i + 1) & 1, kk + 32);                 // overlap next DMA
      asm volatile("s_wait_asynccnt 0x2" ::: "memory");   // current tile landed
    } else {
      asm volatile("s_wait_asynccnt 0x0" ::: "memory");
    }
    __syncthreads();   // make all waves' DMA portions visible

    v16bf a0 = load_frag(arow0 + kk, kh);
    v16bf a1 = load_frag(arow1 + kk, kh);
#pragma unroll
    for (int t = 0; t < 4; ++t) {
      v16bf b = load_frag(&bshm[i & 1][t * 16 + lm][0], kh);
      acc[t]     = wmma_bf16(a0, b, acc[t]);
      acc[4 + t] = wmma_bf16(a1, b, acc[4 + t]);
    }
    __syncthreads();   // reads done before this buffer is overwritten
  }

  const int batch = m0 / S_LEN;
#pragma unroll
  for (int ms = 0; ms < 2; ++ms) {
    const int s0 = (m0 % S_LEN) + ms * 16 + kh * 8;   // seq pos of elem j is s0+j
#pragma unroll
    for (int t = 0; t < 4; ++t) {
      const v8f a = acc[ms * 4 + t];
      const int o = n0 + t * 16 + lm;                 // output channel (this lane)
      const int h = o >> 6, d = o & 63;
      if (which == 2) {
        v8bf pk;                                      // V^T: 8 consecutive s -> 16B store
#pragma unroll
        for (int j = 0; j < 8; ++j) pk[j] = (__bf16)a[j];
        *(v8bf*)(Vtb + ((size_t)(batch * NHEADS + h) * DK + d) * S_LEN + s0) = pk;
      } else {
        const float scale = (which == 0) ? 0.125f : 1.0f;  // fold 1/sqrt(dk) into Q
        __bf16* dst = ((which == 0) ? Qb : Kb) +
                      ((size_t)(batch * NHEADS + h) * S_LEN + s0) * DK + d;
#pragma unroll
        for (int j = 0; j < 8; ++j) dst[(size_t)j * DK] = (__bf16)(a[j] * scale);
      }
    }
  }
}

// ---------------------------------------------------------------------------
// Flash attention (causal, online softmax). grid = (b*h, s/64), block = 128.
// Each wave owns 16 query rows and loops over 32-wide key tiles up to the
// diagonal.  K/V tiles are async-DMA'd into per-wave double-buffered LDS
// (no block barriers -- waves have divergent causal trip counts; per-wave
// s_wait_asynccnt is sufficient since each wave reads only its own tiles).
// ---------------------------------------------------------------------------
__global__ void __launch_bounds__(128)
attn_kernel(const __bf16* __restrict__ Qb,
            const __bf16* __restrict__ Kb,
            const __bf16* __restrict__ Vtb,
            __bf16* __restrict__ AOb) {
  __shared__ __align__(16) __bf16 kshm[4][2][32][64];  // [wave][buf][kpos][dk] 32KB
  __shared__ __align__(16) __bf16 vshm[4][2][64][32];  // [wave][buf][dk][kpos] 32KB
  __shared__ __align__(16) __bf16 plds[4][16][32];     // per-wave P tile      4KB

  const int lane = threadIdx.x & 31;
  const int wid  = threadIdx.x >> 5;
  const int kh   = lane >> 4;
  const int lm   = lane & 15;

  const int bh    = blockIdx.x;                       // 0..23
  const int m0    = blockIdx.y * 64 + wid * 16;       // q row within sequence
  const int batch = bh / NHEADS;
  const int head  = bh % NHEADS;

  const __bf16* Qp = Qb  + (size_t)bh * S_LEN * DK;
  const __bf16* Kp = Kb  + (size_t)bh * S_LEN * DK;
  const __bf16* Vp = Vtb + (size_t)bh * DK * S_LEN;

  // Per-wave async DMA of one 32-wide kpos tile of K (32x64) and V^T (64x32):
  // 2 * 256 chunks of 16B, 32 lanes -> 16 async instructions per tile.
  auto issue_kv = [&](int buf, int kb) {
#pragma unroll
    for (int i = 0; i < 8; ++i) {            // K tile: row=kpos, 8 chunks/row
      const int r = i * 4 + (lane >> 3), p = lane & 7;
      async_copy_b128(Kp + (size_t)(kb + r) * DK + p * 8, &kshm[wid][buf][r][p * 8]);
    }
#pragma unroll
    for (int i = 0; i < 8; ++i) {            // V tile: row=dk, 4 chunks/row
      const int r = i * 8 + (lane >> 2), p = lane & 3;
      async_copy_b128(Vp + (size_t)r * S_LEN + kb + p * 8, &vshm[wid][buf][r][p * 8]);
    }
  };

  v16bf aq0 = load_frag(Qp + (size_t)(m0 + lm) * DK +  0, kh);
  v16bf aq1 = load_frag(Qp + (size_t)(m0 + lm) * DK + 32, kh);

  v8f vo[4] = {};
  float mrow[8], lrow[8];
#pragma unroll
  for (int j = 0; j < 8; ++j) { mrow[j] = -INFINITY; lrow[j] = 0.0f; }

  const int row0 = m0 + kh * 8;        // global q row of acc element j is row0+j
  const int kend = m0 + 16;            // exclusive causal kpos limit

  issue_kv(0, 0);
#pragma unroll 1   // keep single body: accumulators stay pinned across steps
  for (int kb = 0; kb < kend; kb += 32) {
    const int cur = (kb >> 5) & 1;
    if (kb + 32 < kend) {
      issue_kv(cur ^ 1, kb + 32);                         // overlap next DMA
      asm volatile("s_wait_asynccnt 0x10" ::: "memory");  // current tile landed
    } else {
      asm volatile("s_wait_asynccnt 0x0" ::: "memory");
    }

    // ---- scores: S = (Q/8) . K^T, two 16-wide column tiles ----
    v8f s0 = {}, s1 = {};
    {
      v16bf bk;
      bk = load_frag(&kshm[wid][cur][lm][0],       kh); s0 = wmma_bf16(aq0, bk, s0);
      bk = load_frag(&kshm[wid][cur][lm][32],      kh); s0 = wmma_bf16(aq1, bk, s0);
      bk = load_frag(&kshm[wid][cur][16 + lm][0],  kh); s1 = wmma_bf16(aq0, bk, s1);
      bk = load_frag(&kshm[wid][cur][16 + lm][32], kh); s1 = wmma_bf16(aq1, bk, s1);
    }
    // ---- causal mask + per-row tile max ----
    float tmax[8];
#pragma unroll
    for (int j = 0; j < 8; ++j) {
      const int row = row0 + j;
      if (kb + lm > row)      s0[j] = -INFINITY;
      if (kb + 16 + lm > row) s1[j] = -INFINITY;
      tmax[j] = fmaxf(s0[j], s1[j]);
    }
#pragma unroll
    for (int m = 1; m < 16; m <<= 1) {
#pragma unroll
      for (int j = 0; j < 8; ++j)
        tmax[j] = fmaxf(tmax[j], __shfl_xor(tmax[j], m, 32));
    }
    // ---- online softmax update ----
    float corr[8], psum[8];
#pragma unroll
    for (int j = 0; j < 8; ++j) {
      const float nm = fmaxf(mrow[j], tmax[j]);
      corr[j] = __expf(mrow[j] - nm);
      mrow[j] = nm;
      s0[j] = __expf(s0[j] - nm);
      s1[j] = __expf(s1[j] - nm);
      psum[j] = s0[j] + s1[j];
    }
#pragma unroll
    for (int m = 1; m < 16; m <<= 1) {
#pragma unroll
      for (int j = 0; j < 8; ++j)
        psum[j] += __shfl_xor(psum[j], m, 32);
    }
#pragma unroll
    for (int j = 0; j < 8; ++j) lrow[j] = lrow[j] * corr[j] + psum[j];
#pragma unroll
    for (int t = 0; t < 4; ++t)
#pragma unroll
      for (int j = 0; j < 8; ++j) vo[t][j] *= corr[j];

    // ---- relayout P: D-layout -> A-layout via LDS (bf16) ----
#pragma unroll
    for (int j = 0; j < 8; ++j) {
      plds[wid][kh * 8 + j][lm]      = (__bf16)s0[j];
      plds[wid][kh * 8 + j][16 + lm] = (__bf16)s1[j];
    }
    asm volatile("s_wait_dscnt 0x0" ::: "memory");   // drain intra-wave LDS stores
    v16bf ap = load_frag(&plds[wid][lm][0], kh);

    // ---- O += P . V  (V^T tile: row = dk, contiguous in kpos) ----
#pragma unroll
    for (int t = 0; t < 4; ++t) {
      v16bf bv = load_frag(&vshm[wid][cur][t * 16 + lm][0], kh);
      vo[t] = wmma_bf16(ap, bv, vo[t]);
    }
  }

  // ---- epilogue: normalize and store attn_out as bf16 (b, s, h*dk) ----
  float rinv[8];
#pragma unroll
  for (int j = 0; j < 8; ++j) rinv[j] = 1.0f / lrow[j];
#pragma unroll
  for (int t = 0; t < 4; ++t) {
    const int col = head * DK + t * 16 + lm;
    __bf16* dst = AOb + ((size_t)(batch * S_LEN) + row0) * DMODEL + col;
#pragma unroll
    for (int j = 0; j < 8; ++j)
      dst[(size_t)j * DMODEL] = (__bf16)(vo[t][j] * rinv[j]);
  }
}

// ---------------------------------------------------------------------------
// Output projection: out = attn_out @ Wo^T -> fp32.
// Same async-LDS double-buffered pipeline. grid=(M/128, N/64), block=128.
// ---------------------------------------------------------------------------
__global__ void __launch_bounds__(128)
out_proj_kernel(const __bf16* __restrict__ Ab,
                const __bf16* __restrict__ Wo,
                float* __restrict__ Out) {
  __shared__ __align__(16) __bf16 bshm[2][64][32];

  const int tid  = threadIdx.x;
  const int lane = tid & 31;
  const int wid  = tid >> 5;
  const int kh   = lane >> 4;
  const int lm   = lane & 15;

  const int m0 = blockIdx.x * 128 + wid * 32;
  const int n0 = blockIdx.y * 64;

  const int r0 = tid >> 2,         p0 = tid & 3;
  const int r1 = (tid + 128) >> 2, p1 = tid & 3;
  auto issue_b_copy = [&](int buf, int kk) {
    async_copy_b128(Wo + (size_t)(n0 + r0) * DMODEL + kk + p0 * 8, &bshm[buf][r0][p0 * 8]);
    async_copy_b128(Wo + (size_t)(n0 + r1) * DMODEL + kk + p1 * 8, &bshm[buf][r1][p1 * 8]);
  };

  const __bf16* arow0 = Ab + (size_t)(m0 + lm) * DMODEL;
  const __bf16* arow1 = Ab + (size_t)(m0 + 16 + lm) * DMODEL;

  v8f acc[8] = {};

  issue_b_copy(0, 0);
#pragma unroll 1
  for (int i = 0; i < KSTEPS; ++i) {
    const int kk = i * 32;
    if (i + 1 < KSTEPS) {
      issue_b_copy((i + 1) & 1, kk + 32);
      asm volatile("s_wait_asynccnt 0x2" ::: "memory");
    } else {
      asm volatile("s_wait_asynccnt 0x0" ::: "memory");
    }
    __syncthreads();

    v16bf a0 = load_frag(arow0 + kk, kh);
    v16bf a1 = load_frag(arow1 + kk, kh);
#pragma unroll
    for (int t = 0; t < 4; ++t) {
      v16bf b = load_frag(&bshm[i & 1][t * 16 + lm][0], kh);
      acc[t]     = wmma_bf16(a0, b, acc[t]);
      acc[4 + t] = wmma_bf16(a1, b, acc[4 + t]);
    }
    __syncthreads();
  }

#pragma unroll
  for (int ms = 0; ms < 2; ++ms) {
    const int row0 = m0 + ms * 16 + kh * 8;
#pragma unroll
    for (int t = 0; t < 4; ++t) {
      const int col = n0 + t * 16 + lm;
      float* dst = Out + (size_t)row0 * DMODEL + col;
#pragma unroll
      for (int j = 0; j < 8; ++j) dst[(size_t)j * DMODEL] = acc[ms * 4 + t][j];
    }
  }
}

// ---------------------------------------------------------------------------
extern "C" void kernel_launch(void* const* d_in, const int* in_sizes, int n_in,
                              void* d_out, int out_size, void* d_ws, size_t ws_size,
                              hipStream_t stream) {
  const float* x  = (const float*)d_in[0];
  const float* wq = (const float*)d_in[1];
  const float* wk = (const float*)d_in[2];
  const float* wv = (const float*)d_in[3];
  const float* wo = (const float*)d_in[4];
  float* out = (float*)d_out;

  // workspace carve-up (all bf16, sizes are multiples of 256B)
  char* ws = (char*)d_ws;
  size_t off = 0;
  auto carve = [&](size_t elems) {
    void* p = ws + off;
    off += elems * sizeof(__bf16);
    return (__bf16*)p;
  };
  __bf16* xb  = carve((size_t)NROWS * DMODEL);                // 4096x768
  __bf16* wqb = carve((size_t)DMODEL * DMODEL);
  __bf16* wkb = carve((size_t)DMODEL * DMODEL);
  __bf16* wvb = carve((size_t)DMODEL * DMODEL);
  __bf16* wob = carve((size_t)DMODEL * DMODEL);
  __bf16* Qb  = carve((size_t)NBATCH * NHEADS * S_LEN * DK);  // (b,h,s,dk), pre-scaled
  __bf16* Kb  = carve((size_t)NBATCH * NHEADS * S_LEN * DK);  // (b,h,s,dk)
  __bf16* Vtb = carve((size_t)NBATCH * NHEADS * DK * S_LEN);  // (b,h,dk,s)
  __bf16* AOb = carve((size_t)NROWS * DMODEL);                // (b,s,h*dk)

  // 1) fp32 -> bf16 conversions
  {
    const int nx = NROWS * DMODEL / 4;
    cvt_bf16_kernel<<<(nx + 255) / 256, 256, 0, stream>>>(x, xb, nx);
    const int nw = DMODEL * DMODEL / 4;
    cvt_bf16_kernel<<<(nw + 255) / 256, 256, 0, stream>>>(wq, wqb, nw);
    cvt_bf16_kernel<<<(nw + 255) / 256, 256, 0, stream>>>(wk, wkb, nw);
    cvt_bf16_kernel<<<(nw + 255) / 256, 256, 0, stream>>>(wv, wvb, nw);
    cvt_bf16_kernel<<<(nw + 255) / 256, 256, 0, stream>>>(wo, wob, nw);
  }

  // 2) fused QKV projection (async-LDS pipelined GEMM)
  qkv_proj_kernel<<<dim3(NROWS / 128, DMODEL / 64, 3), 128, 0, stream>>>(
      xb, wqb, wkb, wvb, Qb, Kb, Vtb);

  // 3) causal flash attention (async-LDS K/V pipeline)
  attn_kernel<<<dim3(NBATCH * NHEADS, S_LEN / 64), 128, 0, stream>>>(
      Qb, Kb, Vtb, AOb);

  // 4) output projection (fp32 out)
  out_proj_kernel<<<dim3(NROWS / 128, DMODEL / 64), 128, 0, stream>>>(
      AOb, wob, out);
}
